// HRFormerModule_84765474554702
// MI455X (gfx1250) — matrix operand order repaired and checked by reading, not compiled, not run on hardware
//
#include <hip/hip_runtime.h>
#include <stdint.h>

typedef __attribute__((ext_vector_type(16))) _Float16 v16h;
typedef __attribute__((ext_vector_type(8)))  float    v8f;
typedef __attribute__((ext_vector_type(4)))  int      v4i;

union Frag {
    v16h h;
    uint32_t u[8];
    _Float16 e[16];
};

// ---- CDNA5 async global->LDS path (guarded; sync-copy fallback) ----------
#if defined(__gfx1250__) && __has_builtin(__builtin_amdgcn_global_load_async_to_lds_b128)
typedef __attribute__((address_space(1))) v4i v4i_g;
typedef __attribute__((address_space(3))) v4i v4i_l;
#define ASYNC_COPY_B128(g, l)                                                  \
    __builtin_amdgcn_global_load_async_to_lds_b128(                            \
        (v4i_g*)(size_t)(g), (v4i_l*)(uint32_t)(size_t)(l), 0, 0)
#if __has_builtin(__builtin_amdgcn_s_wait_asynccnt)
#define ASYNC_WAIT(n) __builtin_amdgcn_s_wait_asynccnt(n)
#else
#define ASYNC_WAIT(n) asm volatile("s_wait_asynccnt %0" ::"n"(n) : "memory")
#endif
#else
#define ASYNC_COPY_B128(g, l) (*(uint4*)(l) = *(const uint4*)(g))
#define ASYNC_WAIT(n)
#endif

// -------------------------------------------------------------------------
// Weight packer: f32 row-major [K, Nn] -> f16 fragments in B-operand order.
// Fragment (kt, nt) = 32 lanes x 16 halves; lane L supplies B[k][n] with
// n = nt*16 + (L&15), k = kt*32 + (L>>4)*16 + j  (j = 0..15 contiguous).
// -------------------------------------------------------------------------
__global__ __launch_bounds__(128)
void pack_w(const float* __restrict__ W, _Float16* __restrict__ Wp, int K, int Nn)
{
    const int wave = threadIdx.x >> 5, lane = threadIdx.x & 31;
    const int nl = lane & 15, hi = lane >> 4;
    const int NnT = Nn >> 4;
    const int nfr = (K >> 5) * NnT;
    int frag = blockIdx.x * 4 + wave;
    if (frag >= nfr) return;
    int kt = frag / NnT, nt = frag % NnT;
    _Float16* dst = Wp + (((size_t)frag) << 9) + (size_t)lane * 16;
#pragma unroll
    for (int j = 0; j < 16; ++j) {
        int k = kt * 32 + hi * 16 + j;
        dst[j] = (_Float16)W[(size_t)k * Nn + nt * 16 + nl];
    }
}

// -------------------------------------------------------------------------
// LayerNorm over C=384.
// mode 0: src = x in [B,C,H,W] f32, dst = f16 tokens in window order.
// mode 1: src = f32 row-major [P,384] pixel order, dst = f16 same order.
// -------------------------------------------------------------------------
__global__ __launch_bounds__(128)
void ln_kernel(const float* __restrict__ src, const float* __restrict__ g,
               const float* __restrict__ b, _Float16* __restrict__ dst, int mode)
{
    __shared__ float s1[128], s2[128];
    const int p = blockIdx.x;
    const int tid = threadIdx.x;
    float vals[3];
    size_t obase;
    float lsum = 0.f, lsq = 0.f;

    if (mode == 0) {
        int bb = p / 12544, rem = p % 12544;
        int h = rem / 112, wc = rem % 112;
#pragma unroll
        for (int i = 0; i < 3; ++i) {
            int c = tid + i * 128;
            float v = src[(((size_t)bb * 384 + c) * 112 + h) * 112 + wc];
            vals[i] = v; lsum += v; lsq += v * v;
        }
        int widx = bb * 256 + (h / 7) * 16 + (wc / 7);
        int t = (h % 7) * 7 + (wc % 7);
        obase = ((size_t)widx * 49 + t) * 384;
    } else {
        size_t base = (size_t)p * 384;
#pragma unroll
        for (int i = 0; i < 3; ++i) {
            int c = tid + i * 128;
            float v = src[base + c];
            vals[i] = v; lsum += v; lsq += v * v;
        }
        obase = base;
    }
    s1[tid] = lsum; s2[tid] = lsq;
    __syncthreads();
    for (int off = 64; off > 0; off >>= 1) {
        if (tid < off) { s1[tid] += s1[tid + off]; s2[tid] += s2[tid + off]; }
        __syncthreads();
    }
    float mean = s1[0] * (1.f / 384.f);
    float var  = s2[0] * (1.f / 384.f) - mean * mean;
    float rstd = rsqrtf(var + 1e-5f);
#pragma unroll
    for (int i = 0; i < 3; ++i) {
        int c = tid + i * 128;
        dst[obase + c] = (_Float16)((vals[i] - mean) * rstd * g[c] + b[c]);
    }
}

// -------------------------------------------------------------------------
// WMMA GEMM, 64x128 block tile. B double-buffered through LDS via async
// global->LDS copies; A fragment software-pipelined (next tile's A loads
// overlap current tile's 8-WMMA chain). 128 threads = 4 waves.
// EPI: 0 = store f16, 1 = GELU->f16, 2 = proj + shortcut (window reverse,
// f32 pixel-order out), 3 = fc2 + residual, write BCHW f32 to d_out.
// -------------------------------------------------------------------------
template <int EPI>
__global__ __launch_bounds__(128)
void gemm_kernel(const _Float16* __restrict__ A, const _Float16* __restrict__ Wp,
                 const float* __restrict__ bias, int Nn, int K,
                 _Float16* __restrict__ outH, float* __restrict__ outF,
                 const float* __restrict__ aux, float* __restrict__ dOut)
{
    __shared__ __align__(16) _Float16 Bt[2][8 * 512];   // 2 x 8KB B tiles
    const int tid  = threadIdx.x;
    const int wave = tid >> 5, lane = tid & 31;
    const int nl = lane & 15, hi = lane >> 4;
    const int m0 = blockIdx.x * 64 + wave * 16;
    const int n0 = blockIdx.y * 128;
    const int NnT = Nn >> 4;
    const int KT  = K >> 5;
    const int nt0 = n0 >> 4;

    const _Float16* arow = A + (size_t)(m0 + nl) * K;

    // prefetch K-tile 0 of B (8 contiguous fragments = 8KB contiguous)
    {
        const _Float16* src = Wp + (((size_t)nt0) << 9) + tid * 32;
#pragma unroll
        for (int i = 0; i < 4; ++i)
            ASYNC_COPY_B128(src + i * 8, &Bt[0][tid * 32 + i * 8]);
    }
    // prefetch A fragment for tile 0
    Frag a;
    {
        const uint4* ap = (const uint4*)(arow + hi * 16);
        ((uint4*)a.u)[0] = ap[0];
        ((uint4*)a.u)[1] = ap[1];
    }

    v8f acc[8] = {};

    for (int kt = 0; kt < KT; ++kt) {
        const int cur = kt & 1;
        Frag an;
        if (kt + 1 < KT) {
            const _Float16* src =
                Wp + (((size_t)(kt + 1) * NnT + nt0) << 9) + tid * 32;
#pragma unroll
            for (int i = 0; i < 4; ++i)
                ASYNC_COPY_B128(src + i * 8, &Bt[cur ^ 1][tid * 32 + i * 8]);
            // A fragment for next tile overlaps this tile's WMMA chain
            const uint4* ap = (const uint4*)(arow + (kt + 1) * 32 + hi * 16);
            ((uint4*)an.u)[0] = ap[0];
            ((uint4*)an.u)[1] = ap[1];
            ASYNC_WAIT(4);   // 4 newly issued outstanding; tile kt complete
        } else {
            ASYNC_WAIT(0);
        }
        __syncthreads();

#pragma unroll
        for (int s = 0; s < 8; ++s) {
            const uint4* bp = (const uint4*)(&Bt[cur][(s << 9) + lane * 16]);
            Frag b;
            ((uint4*)b.u)[0] = bp[0];
            ((uint4*)b.u)[1] = bp[1];
            acc[s] = __builtin_amdgcn_wmma_f32_16x16x32_f16(
                false, a.h, false, b.h, (short)0, acc[s], false, false);
        }
        __syncthreads();
        a = an;
    }

#pragma unroll
    for (int s = 0; s < 8; ++s) {
        int n = n0 + s * 16 + nl;
        float bv = bias[n];
#pragma unroll
        for (int r = 0; r < 8; ++r) {
            int m = m0 + r + 8 * hi;
            float v = acc[s][r] + bv;
            if (EPI == 0) {
                outH[(size_t)m * Nn + n] = (_Float16)v;
            } else if (EPI == 1) {
                float gl = 0.5f * v * (1.f + erff(v * 0.70710678118f));
                outH[(size_t)m * Nn + n] = (_Float16)gl;
            } else if (EPI == 2) {
                int wdw = m / 49, t = m % 49;
                int bb = wdw >> 8, wr = wdw & 255;
                int h = (wr >> 4) * 7 + t / 7;
                int wc = (wr & 15) * 7 + t % 7;
                size_t pix = ((size_t)bb * 112 + h) * 112 + wc;
                float sc = aux[(((size_t)bb * 384 + n) * 112 + h) * 112 + wc];
                outF[pix * 384 + n] = v + sc;
            } else {
                int bb = m / 12544, rem = m % 12544;
                int h = rem / 112, wc = rem % 112;
                float res = aux[(size_t)m * 384 + n];
                dOut[(((size_t)bb * 384 + n) * 112 + h) * 112 + wc] = v + res;
            }
        }
    }
}

// -------------------------------------------------------------------------
// Windowed attention: one block (4 waves) per (window, head). N=49 pad 64.
// S = Q K^T (WMMA) + rel-pos bias -> masked softmax -> P V (WMMA).
// All fragment loads are batched ahead of each WMMA chain so loads clause
// and WMMAs issue back-to-back (accumulator chaining D->C has no hazard).
// -------------------------------------------------------------------------
__global__ __launch_bounds__(128)
void attn_kernel(const _Float16* __restrict__ qkv, const float* __restrict__ rpb,
                 _Float16* __restrict__ out)
{
    __shared__ __align__(16) _Float16 Plds[4][16 * 64];  // 8KB
    __shared__ __align__(16) _Float16 Vlds[4 * 512];     // 4KB, 4 B-fragments
    const int w = blockIdx.x;
    const int head = blockIdx.y;
    const int tid = threadIdx.x;
    const int wave = tid >> 5, lane = tid & 31;
    const int nl = lane & 15, hi = lane >> 4;
    const int t0 = wave * 16;

    // ---- cooperative V staging: fragment fi = (kk/32)*2 + s2 ----
    {
        int fi = tid >> 5, ln2 = tid & 31;
        int hi2 = ln2 >> 4, nl2 = ln2 & 15;
        int kkb = (fi >> 1) * 32, s2 = fi & 1;
        _Float16* dst = &Vlds[(fi << 9) + ln2 * 16];
#pragma unroll
        for (int j = 0; j < 16; ++j) {
            int kt = kkb + hi2 * 16 + j;
            int tok = w * 49 + (kt < 49 ? kt : 0);
            dst[j] = qkv[(size_t)tok * 1152 + 768 + head * 32 + s2 * 16 + nl2];
        }
    }

    // ---- batched operand loads: Q fragment + all 4 K fragments ----
    Frag aq;
    {
        int row = t0 + nl;
        int tok = w * 49 + (row < 49 ? row : 0);
        const uint4* qp = (const uint4*)(qkv + (size_t)tok * 1152 + head * 32 + hi * 16);
        ((uint4*)aq.u)[0] = qp[0];
        ((uint4*)aq.u)[1] = qp[1];
    }
    Frag bk[4];
#pragma unroll
    for (int s = 0; s < 4; ++s) {
        int col = s * 16 + nl;
        int tok = w * 49 + (col < 49 ? col : 0);
        const uint4* kp =
            (const uint4*)(qkv + (size_t)tok * 1152 + 384 + head * 32 + hi * 16);
        ((uint4*)bk[s].u)[0] = kp[0];
        ((uint4*)bk[s].u)[1] = kp[1];
    }

    // ---- S = Q K^T : 4 back-to-back WMMAs ----
    v8f sacc[4] = {};
#pragma unroll
    for (int s = 0; s < 4; ++s)
        sacc[s] = __builtin_amdgcn_wmma_f32_16x16x32_f16(
            false, aq.h, false, bk[s].h, (short)0, sacc[s], false, false);

    // ---- scale + relative position bias + mask ----
    float pv[4][8];
#pragma unroll
    for (int s = 0; s < 4; ++s) {
        int col = s * 16 + nl;
#pragma unroll
        for (int r = 0; r < 8; ++r) {
            int row = t0 + r + 8 * hi;
            float v = -1e30f;
            if (row < 49 && col < 49) {
                int ri = row / 7, ci = row % 7, rj = col / 7, cj = col % 7;
                int idx = (ri - rj + 6) * 13 + (ci - cj + 6);
                v = sacc[s][r] * 0.17677669529663687f + rpb[idx * 12 + head];
            }
            pv[s][r] = v;
        }
    }

    // ---- row softmax (row spans 16 lanes of same hi, 4 values/lane) ----
#pragma unroll
    for (int r = 0; r < 8; ++r) {
        float mx = fmaxf(fmaxf(pv[0][r], pv[1][r]), fmaxf(pv[2][r], pv[3][r]));
        for (int mk = 8; mk >= 1; mk >>= 1) mx = fmaxf(mx, __shfl_xor(mx, mk, 32));
        float e[4], sum = 0.f;
#pragma unroll
        for (int s = 0; s < 4; ++s) { e[s] = __expf(pv[s][r] - mx); sum += e[s]; }
        for (int mk = 8; mk >= 1; mk >>= 1) sum += __shfl_xor(sum, mk, 32);
        float inv = 1.f / sum;
#pragma unroll
        for (int s = 0; s < 4; ++s)
            Plds[wave][(r + 8 * hi) * 64 + s * 16 + nl] = (_Float16)(e[s] * inv);
    }

    __syncthreads();   // Vlds staging complete for all waves

    // ---- batched operand loads for O = P V ----
    Frag ap[2], vb[4];
#pragma unroll
    for (int t = 0; t < 2; ++t) {
        const uint4* pp =
            (const uint4*)(&Plds[wave][nl * 64 + t * 32 + hi * 16]);
        ((uint4*)ap[t].u)[0] = pp[0];
        ((uint4*)ap[t].u)[1] = pp[1];
#pragma unroll
        for (int s2 = 0; s2 < 2; ++s2) {
            const uint4* vp =
                (const uint4*)(&Vlds[((t * 2 + s2) << 9) + lane * 16]);
            ((uint4*)vb[t * 2 + s2].u)[0] = vp[0];
            ((uint4*)vb[t * 2 + s2].u)[1] = vp[1];
        }
    }

    // ---- O = P V : 4 back-to-back WMMAs (K=64 over 2 steps, N=32) ----
    v8f oacc[2] = {};
#pragma unroll
    for (int t = 0; t < 2; ++t)
#pragma unroll
        for (int s2 = 0; s2 < 2; ++s2)
            oacc[s2] = __builtin_amdgcn_wmma_f32_16x16x32_f16(
                false, ap[t].h, false, vb[t * 2 + s2].h, (short)0,
                oacc[s2], false, false);

#pragma unroll
    for (int s2 = 0; s2 < 2; ++s2) {
#pragma unroll
        for (int r = 0; r < 8; ++r) {
            int row = t0 + r + 8 * hi;
            if (row < 49) {
                int d = s2 * 16 + nl;
                out[((size_t)(w * 49 + row)) * 384 + head * 32 + d] =
                    (_Float16)oacc[s2][r];
            }
        }
    }
}

// -------------------------------------------------------------------------
extern "C" void kernel_launch(void* const* d_in, const int* in_sizes, int n_in,
                              void* d_out, int out_size, void* d_ws, size_t ws_size,
                              hipStream_t stream)
{
    const float* x      = (const float*)d_in[0];
    const float* n1g    = (const float*)d_in[1];
    const float* n1b    = (const float*)d_in[2];
    const float* qkv_w  = (const float*)d_in[3];
    const float* qkv_b  = (const float*)d_in[4];
    const float* rpb    = (const float*)d_in[5];
    const float* proj_w = (const float*)d_in[6];
    const float* proj_b = (const float*)d_in[7];
    const float* n2g    = (const float*)d_in[8];
    const float* n2b    = (const float*)d_in[9];
    const float* fc1_w  = (const float*)d_in[10];
    const float* fc1_b  = (const float*)d_in[11];
    const float* fc2_w  = (const float*)d_in[12];
    const float* fc2_b  = (const float*)d_in[13];
    float* out = (float*)d_out;

    constexpr size_t P = 100352;          // pixels == window tokens
    constexpr size_t S16 = P * 384 * 2;   // one f16 activation plane

    char* ws = (char*)d_ws;
    _Float16* xw   = (_Float16*)(ws);                       // LN1 tokens (f16)
    _Float16* bufC = (_Float16*)(ws + S16);                 // attn_out, then h1
    _Float16* bufB = (_Float16*)(ws + 2 * S16);             // qkv, then fc1 out
    float*    xres = (float*)(ws + 2 * S16 + P * 1536 * 2); // residual carrier
    char* wb = ws + 2 * S16 + P * 1536 * 2 + P * 384 * 4;
    _Float16* qkvP  = (_Float16*)(wb);
    _Float16* projP = (_Float16*)(wb + 384 * 1152 * 2);
    _Float16* fc1P  = (_Float16*)(wb + 384 * 1152 * 2 + 384 * 384 * 2);
    _Float16* fc2P  = (_Float16*)(wb + 384 * 1152 * 2 + 384 * 384 * 2 + 384 * 1536 * 2);

    // pack weights f32 -> f16 B-fragment layout
    pack_w<<<dim3((864 + 3) / 4), 128, 0, stream>>>(qkv_w, qkvP, 384, 1152);
    pack_w<<<dim3((288 + 3) / 4), 128, 0, stream>>>(proj_w, projP, 384, 384);
    pack_w<<<dim3((1152 + 3) / 4), 128, 0, stream>>>(fc1_w, fc1P, 384, 1536);
    pack_w<<<dim3((1152 + 3) / 4), 128, 0, stream>>>(fc2_w, fc2P, 1536, 384);

    // LN1 + window partition
    ln_kernel<<<dim3(100352), 128, 0, stream>>>(x, n1g, n1b, xw, 0);

    // QKV projection: [P,384] x [384,1152]
    gemm_kernel<0><<<dim3(100352 / 64, 1152 / 128), 128, 0, stream>>>(
        xw, qkvP, qkv_b, 1152, 384, bufB, nullptr, nullptr, nullptr);

    // windowed attention (2048 windows x 12 heads)
    attn_kernel<<<dim3(2048, 12), 128, 0, stream>>>(bufB, rpb, bufC);

    // output projection + shortcut residual (window reverse -> pixel order)
    gemm_kernel<2><<<dim3(100352 / 64, 384 / 128), 128, 0, stream>>>(
        bufC, projP, proj_b, 384, 384, nullptr, xres, x, nullptr);

    // LN2 -> h1 (f16, pixel order) into bufC
    ln_kernel<<<dim3(100352), 128, 0, stream>>>(xres, n2g, n2b, bufC, 1);

    // FC1 + exact GELU: [P,384] x [384,1536]
    gemm_kernel<1><<<dim3(100352 / 64, 1536 / 128), 128, 0, stream>>>(
        bufC, fc1P, fc1_b, 1536, 384, bufB, nullptr, nullptr, nullptr);

    // FC2 + residual, final BHWC -> BCHW transpose into d_out
    gemm_kernel<3><<<dim3(100352 / 64, 384 / 128), 128, 0, stream>>>(
        bufB, fc2P, fc2_b, 384, 1536, nullptr, nullptr, xres, out);
}
